// kspaceMap_84911503442466
// MI455X (gfx1250) — compile-verified, matrix operand order checked by reading, not run on hardware
//
#include <hip/hip_runtime.h>
#include <hip/hip_fp16.h>

// kspaceMap forward, algebraically reduced:
//   sep[b,c,m,n] = |FFT_H(img_b)[c,n]| / sqrt(128)   (independent of m)
//   emb[b,o,m,n] = lrelu( sum_c W[o,c]*sep[b,c,·,n] )  (broadcast over m)
//   out = concat(emb, x[:,2:]) -> (8,129,128,128)
// MI455X roofline: compute ~170 MFLOP total (negligible); bound by the
// 67.6 MB broadcast store (~3 us at 23.3 TB/s). GEMM done with
// V_WMMA_F32_16X16X4_F32 (full f32 precision, free at this size).

#define IM 128
#define LRELU 0.2f

typedef __attribute__((ext_vector_type(2)))  float    v2f;
typedef __attribute__((ext_vector_type(8)))  float    v8f;
typedef __attribute__((ext_vector_type(16))) _Float16 v16h;

// ---------------------------------------------------------------------------
// Kernel A: per (b, n) column, 128-point DFT along H via LDS twiddle table.
// Aout[b,c,n] = | sum_m img[b,m,n] * e^{-2*pi*i*c*m/128} | / 128
// (ortho FFT 1/sqrt(128) times the extra 1/sqrt(128) from |ifft2| phase row)
// ---------------------------------------------------------------------------
__global__ void dft_mag_kernel(const float* __restrict__ x,
                               float* __restrict__ Aout) {
  const int n = blockIdx.x;   // column
  const int b = blockIdx.y;   // batch
  const int c = threadIdx.x;  // output frequency, 0..127
  __shared__ float re[IM], im[IM], twc[IM], tws[IM];

  re[c] = x[(((size_t)b * 3 + 0) * IM + c) * IM + n];
  im[c] = x[(((size_t)b * 3 + 1) * IM + c) * IM + n];
  float ang = (2.0f * 3.14159265358979323846f / (float)IM) * (float)c;
  float s, co;
  __sincosf(ang, &s, &co);
  twc[c] = co;          // e^{-i*ang} = cos - i*sin  (sign matters: G[c] vs G[-c])
  tws[c] = -s;
  __syncthreads();

  float ar = 0.f, ai = 0.f;
#pragma unroll 4
  for (int m = 0; m < IM; ++m) {
    int k = (c * m) & (IM - 1);
    float tc = twc[k], ts = tws[k];
    float r = re[m], q = im[m];
    ar = fmaf(r, tc, ar); ar = fmaf(-q, ts, ar);   // re*tc - im*ts
    ai = fmaf(r, ts, ai); ai = fmaf( q, tc, ai);   // re*ts + im*tc
  }
  Aout[((size_t)b * IM + c) * IM + n] = sqrtf(ar * ar + ai * ai) * (1.0f / 128.0f);
}

// ---------------------------------------------------------------------------
// Kernel B: M_b = lrelu( W (128x128) * A_b (128x128) ), one wave per 16x16
// tile, f32 WMMA (16x16x4), K-loop of 32.
// Fragment layouts per ISA 7.12.2:
//   A 16x4 f32:  lane&15 = M row; (lane>>4)*2 = K half-offset; v0/v1 = K,K+1
//   B 4x16 f32:  lane&15 = N col; same half-wave K split (B rows across lanes)
//   C/D 16x16:   VGPR r: lanes0-15 -> M=r, lanes16-31 -> M=r+8; N = lane&15
// ---------------------------------------------------------------------------
__global__ void wmma_gemm_lrelu_kernel(const float* __restrict__ W,
                                       const float* __restrict__ Ain,
                                       float* __restrict__ Mout) {
  const int ntile = blockIdx.x * 16;
  const int otile = blockIdx.y * 16;
  const int b     = blockIdx.z;
  const int lane  = threadIdx.x;   // wave32 block
  const int rc    = lane & 15;     // M row (A) / N col (B,D)
  const float* Ab = Ain + (size_t)b * IM * IM;

  v8f acc = {};
#if __has_builtin(__builtin_amdgcn_wmma_f32_16x16x4_f32)
  const int koff = (lane >> 4) * 2;
  for (int k = 0; k < IM; k += 4) {
    v2f a, bv;
    a.x  = W[(size_t)(otile + rc) * IM + (k + koff)];
    a.y  = W[(size_t)(otile + rc) * IM + (k + koff + 1)];
    bv.x = Ab[(size_t)(k + koff)     * IM + (ntile + rc)];
    bv.y = Ab[(size_t)(k + koff + 1) * IM + (ntile + rc)];
    acc = __builtin_amdgcn_wmma_f32_16x16x4_f32(
        /*neg_a=*/false, a, /*neg_b=*/false, bv,
        /*c_mod=*/(short)0, acc, /*reuse_a=*/false, /*reuse_b=*/false);
  }
#else
  // Fallback: probe-confirmed f16 WMMA (16x16x32). A 16x32 f16 layout:
  // lane&15 = M, kbase = (lane>>4)*8; element e -> K = (e>>3)*16 + kbase + (e&7)
  const int kbase = (lane >> 4) * 8;
  for (int k = 0; k < IM; k += 32) {
    v16h a, bv;
#pragma unroll
    for (int e = 0; e < 16; ++e) {
      int kk = k + ((e >> 3) * 16) + kbase + (e & 7);
      a[e]  = (_Float16)W[(size_t)(otile + rc) * IM + kk];
      bv[e] = (_Float16)Ab[(size_t)kk * IM + (ntile + rc)];
    }
    acc = __builtin_amdgcn_wmma_f32_16x16x32_f16(
        false, a, false, bv, (short)0, acc, false, false);
  }
#endif

  const int nIdx  = ntile + rc;
  const int mbase = otile + (lane >> 4) * 8;
#pragma unroll
  for (int r = 0; r < 8; ++r) {
    float v = acc[r];
    v = fmaxf(v, LRELU * v);   // LeakyReLU(0.2)
    Mout[((size_t)b * IM + (mbase + r)) * IM + nIdx] = v;
  }
}

// ---------------------------------------------------------------------------
// Kernel C: bandwidth-bound broadcast. out[b,ch,m,n] = M[b,ch,n] for ch<128,
// and out[b,128,m,n] = x[b,2,m,n]. float4 per lane -> 512 B per wave store.
// M rows are read 128x each and stay L1/L2 resident.
// ---------------------------------------------------------------------------
__global__ void broadcast_kernel(const float* __restrict__ M,
                                 const float* __restrict__ x,
                                 float* __restrict__ out) {
  size_t i = (size_t)blockIdx.x * blockDim.x + threadIdx.x;
  const size_t total4 = (size_t)8 * 129 * IM * (IM / 4);
  if (i >= total4) return;
  int n4 = (int)(i & 31);          // float4 index along n
  int m  = (int)((i >> 5) & 127);  // row (broadcast dim)
  size_t t = i >> 12;              // 32 * 128 = 4096
  int ch = (int)(t % 129);
  int b  = (int)(t / 129);
  float4 v;
  if (ch < IM) {
    v = *(const float4*)(M + ((size_t)b * IM + ch) * IM + n4 * 4);
  } else {
    v = *(const float4*)(x + (((size_t)b * 3 + 2) * IM + m) * IM + n4 * 4);
  }
  *(float4*)(out + (((size_t)b * 129 + ch) * IM + m) * IM + n4 * 4) = v;
}

// ---------------------------------------------------------------------------
extern "C" void kernel_launch(void* const* d_in, const int* in_sizes, int n_in,
                              void* d_out, int out_size, void* d_ws, size_t ws_size,
                              hipStream_t stream) {
  const float* x      = (const float*)d_in[0];  // (8,3,128,128)
  // d_in[1] = mask: unused by the reference forward
  const float* conv_w = (const float*)d_in[2];  // (128,128)
  float* out = (float*)d_out;                   // (8,129,128,128)

  float* Aws = (float*)d_ws;                    // |FFT_H| magnitudes: 8*128*128 f32
  float* Mws = Aws + (size_t)8 * IM * IM;       // lrelu(W*A):          8*128*128 f32

  dft_mag_kernel<<<dim3(IM, 8), IM, 0, stream>>>(x, Aws);
  wmma_gemm_lrelu_kernel<<<dim3(IM / 16, IM / 16, 8), 32, 0, stream>>>(conv_w, Aws, Mws);

  size_t total4 = (size_t)8 * 129 * IM * (IM / 4);
  broadcast_kernel<<<(unsigned)((total4 + 255) / 256), 256, 0, stream>>>(Mws, x, out);
}
